// Attention_44702019616803
// MI455X (gfx1250) — matrix operand order repaired and checked by reading, not compiled, not run on hardware
//
#include <hip/hip_runtime.h>

// ---- problem constants -----------------------------------------------------
#define BB    8
#define NN    577
#define DD    768
#define HD    48
#define C3    2304
#define NPAD  608           // 577 padded to multiple of 32 (19 K-chunks)
#define MTL   37            // ceil(577/16) row tiles
#define NROWP 592           // 577 padded to multiple of 16 (padded probs2 rows)
#define M1    (BB*NN)       // 4616
#define M2    (2*M1)        // 9232 (x and x_freq stacked)

typedef __bf16 bhalf;
typedef __bf16 v16bf __attribute__((ext_vector_type(16)));
typedef float  v8f   __attribute__((ext_vector_type(8)));

// ---- WMMA fragment loaders (16x16x32 bf16, contiguous k order shared by A and B) -------------------
// A matrix 16x32: lane M = lane&15; VGPR j packs K pair:
//   k = 2*j + (lane>>4)*16
__device__ inline v16bf frag_a(const bhalf* p, int ld) {
  int lane = threadIdx.x & 31;
  int r = lane & 15, hf = lane >> 4;
  v16bf f;
#pragma unroll
  for (int j = 0; j < 8; ++j) {
    int k = (j << 1) + hf * 16;
    f[2 * j]     = p[r * ld + k];
    f[2 * j + 1] = p[r * ld + k + 1];
  }
  return f;
}
// B matrix 32x16: lane N = lane&15; lanes 0-15 hold K=0..15, lanes 16-31 K=16..31
//   k = 2*j + (lane>>4)*16   (tile stored [n][k] row-major)
__device__ inline v16bf frag_b(const bhalf* p, int ld) {
  int lane = threadIdx.x & 31;
  int r = lane & 15, hf = lane >> 4;
  v16bf f;
#pragma unroll
  for (int j = 0; j < 8; ++j) {
    int k = (j << 1) + hf * 16;
    f[2 * j]     = p[r * ld + k];
    f[2 * j + 1] = p[r * ld + k + 1];
  }
  return f;
}

// ---- f32 -> bf16 conversion ------------------------------------------------
__global__ void f32_to_bf16(const float* __restrict__ s, bhalf* __restrict__ d, int n) {
  int i = blockIdx.x * blockDim.x + threadIdx.x;
  int stride = gridDim.x * blockDim.x;
  for (; i < n; i += stride) d[i] = (bhalf)s[i];
}

// ---- generic WMMA GEMM: C[M,Nout] = A[M,K] @ W[Nout,K]^T + bias ------------
// block 256 = 8 waves; tile 128x64; each wave owns a 16x64 strip (4 acc tiles)
template <bool OUT_BF16>
__global__ __launch_bounds__(256) void gemm_wmma(
    const bhalf* __restrict__ A, int lda, int M,
    const bhalf* __restrict__ W, int K,
    const float* __restrict__ bias,
    void* __restrict__ Cout, int ldc) {
  __shared__ __align__(16) bhalf As[128 * 40];
  __shared__ __align__(16) bhalf Bs[64 * 40];
  const int t = threadIdx.x;
  const int lane = t & 31, wave = t >> 5, hf = lane >> 4;
  const int m0 = blockIdx.x * 128;
  const int n0 = blockIdx.y * 64;
  v8f acc[4] = {};

  for (int k0 = 0; k0 < K; k0 += 32) {
    {  // A tile 128x32: 2 threads per row, 2x uint4 each
      int row = t >> 1, seg = t & 1;
      int gm = m0 + row;
      const uint4* src = (const uint4*)(A + (size_t)gm * lda + k0 + seg * 16);
      uint4 z = {0u, 0u, 0u, 0u};
      uint4 d0 = (gm < M) ? src[0] : z;
      uint4 d1 = (gm < M) ? src[1] : z;
      *(uint4*)(&As[row * 40 + seg * 16])     = d0;
      *(uint4*)(&As[row * 40 + seg * 16 + 8]) = d1;
      if (k0 + 32 < K && gm < M)
        __builtin_prefetch(A + (size_t)gm * lda + k0 + 32 + seg * 16, 0, 1);
    }
    {  // B tile 64x32 from W[n][k]
      int row = t >> 2, seg = t & 3;
      const uint4* src = (const uint4*)(W + (size_t)(n0 + row) * K + k0 + seg * 8);
      *(uint4*)(&Bs[row * 40 + seg * 8]) = src[0];
      if (k0 + 32 < K)
        __builtin_prefetch(W + (size_t)(n0 + row) * K + k0 + 32 + seg * 8, 0, 1);
    }
    __syncthreads();
    v16bf af = frag_a(As + wave * 16 * 40, 40);
#pragma unroll
    for (int tt = 0; tt < 4; ++tt) {
      v16bf bf = frag_b(Bs + tt * 16 * 40, 40);
      acc[tt] = __builtin_amdgcn_wmma_f32_16x16x32_bf16(
          false, af, false, bf, (short)0, acc[tt], false, false);
    }
    __syncthreads();
  }
#pragma unroll
  for (int tt = 0; tt < 4; ++tt)
#pragma unroll
    for (int r = 0; r < 8; ++r) {
      int mrow = m0 + wave * 16 + r + hf * 8;
      int ncol = n0 + tt * 16 + (lane & 15);
      if (mrow < M) {
        float v = acc[tt][r] + bias[ncol];
        if (OUT_BF16) ((bhalf*)Cout)[(size_t)mrow * ldc + ncol] = (bhalf)v;
        else          ((float*)Cout)[(size_t)mrow * ldc + ncol] = v;
      }
    }
}

// ---- scores + first softmax: one block per (channel c, 16-row tile) --------
// c<16: q,k from x; c>=16: from x_freq.  attn[c][n][m] f32, cols padded to 608.
// K tiles staged through LDS with coalesced zero-padded uint4 fills; uniform
// tile loop so __syncthreads() separates fill and WMMA phases.
__global__ __launch_bounds__(128) void score_softmax(
    const bhalf* __restrict__ qkv, float* __restrict__ attn, int b) {
  __shared__ __align__(16) bhalf Qs[16 * 64];      // [r][d], d padded to 64
  __shared__ __align__(16) bhalf Ks[4 * 16 * 64];  // 4 waves x [m][d]
  __shared__ float S[16 * 616];
  __shared__ float tred[16 * 8];
  __shared__ float rmax[16], rsum[16];
  const int t = threadIdx.x;
  const int lane = t & 31, wave = t >> 5, hf = lane >> 4;
  const int c = blockIdx.x;  // 0..31
  const int src = c >> 4, h = c & 15;
  const int n0 = blockIdx.y * 16;
  const size_t rowbase = (size_t)(src * M1 + b * NN);
  const uint4 z4 = {0u, 0u, 0u, 0u};

  {  // Q tile [16][64]: one uint4 per thread, zero-padded (d>=48 or n>=NN)
    int r = t >> 3, seg = t & 7;
    int n = n0 + r;
    uint4 v = z4;
    if (seg < 6 && n < NN)
      v = *(const uint4*)(qkv + (rowbase + n) * C3 + h * HD + seg * 8);
    *(uint4*)(&Qs[r * 64 + seg * 8]) = v;
  }
  __syncthreads();
  v16bf a0 = frag_a(Qs, 64);
  v16bf a1 = frag_a(Qs + 32, 64);
  const float scale = 0.14433756729740645f;  // 1/sqrt(48)

  for (int it = 0; it < (MTL + 3) / 4; ++it) {
    // fill 4 K tiles (one per wave), 4 uint4 per thread, zero-padded
#pragma unroll
    for (int id = t; id < 512; id += 128) {
      int tw = id >> 7, rem = id & 127;
      int r = rem >> 3, seg = rem & 7;
      int m = (it * 4 + tw) * 16 + r;
      uint4 v = z4;
      if (seg < 6 && m < NN)
        v = *(const uint4*)(qkv + (rowbase + m) * C3 + DD + h * HD + seg * 8);
      *(uint4*)(&Ks[(tw * 16 + r) * 64 + seg * 8]) = v;
    }
    __syncthreads();
    int mt = it * 4 + wave;
    if (mt < MTL) {
      const bhalf* kb = Ks + wave * 16 * 64;
      v16bf b0 = frag_b(kb, 64);
      v16bf b1 = frag_b(kb + 32, 64);
      v8f acc = {};
      acc = __builtin_amdgcn_wmma_f32_16x16x32_bf16(false, a0, false, b0, (short)0, acc, false, false);
      acc = __builtin_amdgcn_wmma_f32_16x16x32_bf16(false, a1, false, b1, (short)0, acc, false, false);
#pragma unroll
      for (int r = 0; r < 8; ++r)
        S[(r + hf * 8) * 616 + mt * 16 + (lane & 15)] = acc[r] * scale;
    }
    __syncthreads();
  }

  // row softmax: 8 threads per row
  int row = t & 15, g = t >> 4;
  float mx = -1e30f;
  for (int m = g; m < NN; m += 8) mx = fmaxf(mx, S[row * 616 + m]);
  tred[row * 8 + g] = mx;
  __syncthreads();
  if (g == 0) {
    float v = tred[row * 8];
    for (int i = 1; i < 8; ++i) v = fmaxf(v, tred[row * 8 + i]);
    rmax[row] = v;
  }
  __syncthreads();
  float sm = 0.f;
  for (int m = g; m < NN; m += 8) {
    float e = __expf(S[row * 616 + m] - rmax[row]);
    S[row * 616 + m] = e;
    sm += e;
  }
  tred[row * 8 + g] = sm;
  __syncthreads();
  if (g == 0) {
    float v = 0.f;
    for (int i = 0; i < 8; ++i) v += tred[row * 8 + i];
    rsum[row] = v;
  }
  __syncthreads();
  int n = n0 + row;
  if (n < NN) {
    float inv = 1.0f / rsum[row];
    float* out = attn + ((size_t)c * NN + n) * NPAD;
    for (int m = g; m < NPAD; m += 8)
      out[m] = (m < NN) ? S[row * 616 + m] * inv : 0.0f;
  }
}

// ---- channel mix (1x1 conv across 32 maps) + second softmax ----------------
// one block per output row n; all 32 channel rows live in dynamic LDS (~120KB)
__global__ __launch_bounds__(256) void mix_softmax(
    const float* __restrict__ attn, const float* __restrict__ conv_w,
    const float* __restrict__ conv_b, bhalf* __restrict__ probs2) {
  extern __shared__ float smem[];
  float* Sc = smem;             // 32*608
  float* Mx = Sc + 32 * NPAD;   // 16*608
  float* Wm = Mx + 16 * NPAD;   // 512
  float* Tr = Wm + 512;         // 16*16
  float* Rm = Tr + 256;         // 16
  float* Rs = Rm + 16;          // 16
  const int t = threadIdx.x;
  const int n = blockIdx.x;

  for (int e = t; e < 32 * (NPAD / 4); e += 256) {
    int c = e / (NPAD / 4), i = e % (NPAD / 4);
    ((float4*)Sc)[c * (NPAD / 4) + i] =
        ((const float4*)(attn + ((size_t)c * NN + n) * NPAD))[i];
  }
  for (int e = t; e < 512; e += 256) Wm[e] = conv_w[e];
  __syncthreads();

  for (int o = 0; o < 16; ++o) {
    float bo = conv_b[o];
    for (int m = t; m < NPAD; m += 256) {
      float acc = bo;
#pragma unroll
      for (int c = 0; c < 32; ++c) acc += Wm[o * 32 + c] * Sc[c * NPAD + m];
      Mx[o * NPAD + m] = acc;
    }
  }
  __syncthreads();

  // softmax over m for each of 16 output channels; 16 threads per row
  int row = t >> 4, g = t & 15;
  float mx = -1e30f;
  for (int m = g; m < NN; m += 16) mx = fmaxf(mx, Mx[row * NPAD + m]);
  Tr[row * 16 + g] = mx;
  __syncthreads();
  if (g == 0) {
    float v = Tr[row * 16];
    for (int i = 1; i < 16; ++i) v = fmaxf(v, Tr[row * 16 + i]);
    Rm[row] = v;
  }
  __syncthreads();
  float s = 0.f;
  for (int m = g; m < NN; m += 16) {
    float e = __expf(Mx[row * NPAD + m] - Rm[row]);
    Mx[row * NPAD + m] = e;
    s += e;
  }
  Tr[row * 16 + g] = s;
  __syncthreads();
  if (g == 0) {
    float v = 0.f;
    for (int i = 0; i < 16; ++i) v += Tr[row * 16 + i];
    Rs[row] = v;
  }
  __syncthreads();
  float inv = 1.0f / Rs[row];
  bhalf* out = probs2 + ((size_t)row * NROWP + n) * NPAD;
  for (int m = g; m < NPAD; m += 16)
    out[m] = (bhalf)((m < NN) ? Mx[row * NPAD + m] * inv : 0.0f);
}

// ---- weighted = probs2 @ V : one wave per (head, 16-row tile) --------------
// V tile kept row-major [m][d] in LDS (coalesced uint4 fill); B-fragment is a
// strided u16 gather from LDS.  probs2 rows are padded to NROWP so A-fragment
// loads need no bounds checks.
__global__ __launch_bounds__(32) void weighted_gemm(
    const bhalf* __restrict__ probs2, const bhalf* __restrict__ qkv,
    bhalf* __restrict__ weighted, int b) {
  __shared__ __align__(16) bhalf Vs[32 * 48];  // [m][d]
  const int lane = threadIdx.x & 31;
  const int hf = lane >> 4;
  const int h = blockIdx.x;
  const int n0 = blockIdx.y * 16;
  const size_t vbase = (size_t)(b * NN);
  v8f acc[3] = {};

  for (int k0 = 0; k0 < NPAD; k0 += 32) {
    int m = k0 + lane;
    {  // V tile fill: one row per lane, 6x uint4
      const uint4* vp = (const uint4*)(qkv + (vbase + (m < NN ? m : 0)) * C3 + 2 * DD + h * HD);
      bool valid = (m < NN);
      uint4 z = {0u, 0u, 0u, 0u};
#pragma unroll
      for (int seg = 0; seg < 6; ++seg) {
        uint4 v = valid ? vp[seg] : z;
        *(uint4*)(&Vs[lane * 48 + seg * 8]) = v;
      }
    }
    __syncthreads();
    // A fragment straight from padded probs2 (no guards needed)
    v16bf af;
    {
      int nrow = n0 + (lane & 15);
      const bhalf* ap = probs2 + ((size_t)h * NROWP + nrow) * NPAD + k0;
#pragma unroll
      for (int j = 0; j < 8; ++j) {
        int k = (j << 1) + hf * 16;
        af[2 * j]     = ap[k];
        af[2 * j + 1] = ap[k + 1];
      }
    }
#pragma unroll
    for (int tt = 0; tt < 3; ++tt) {
      int dcol = tt * 16 + (lane & 15);
      v16bf bfb;
#pragma unroll
      for (int j = 0; j < 8; ++j) {
        int k = (j << 1) + hf * 16;
        bfb[2 * j]     = Vs[k * 48 + dcol];
        bfb[2 * j + 1] = Vs[(k + 1) * 48 + dcol];
      }
      acc[tt] = __builtin_amdgcn_wmma_f32_16x16x32_bf16(
          false, af, false, bfb, (short)0, acc[tt], false, false);
    }
    __syncthreads();
  }
#pragma unroll
  for (int tt = 0; tt < 3; ++tt)
#pragma unroll
    for (int r = 0; r < 8; ++r) {
      int nrow = n0 + r + hf * 8;
      int d = tt * 16 + (lane & 15);
      if (nrow < NN)
        weighted[((size_t)(b * NN + nrow)) * DD + h * HD + d] = (bhalf)acc[tt][r];
    }
}

// ---- host orchestration ----------------------------------------------------
extern "C" void kernel_launch(void* const* d_in, const int* in_sizes, int n_in,
                              void* d_out, int out_size, void* d_ws, size_t ws_size,
                              hipStream_t stream) {
  (void)in_sizes; (void)n_in; (void)out_size; (void)ws_size;
  const float* x      = (const float*)d_in[0];
  const float* xf     = (const float*)d_in[1];
  const float* qkv_w  = (const float*)d_in[2];
  const float* qkv_b  = (const float*)d_in[3];
  const float* conv_w = (const float*)d_in[4];
  const float* conv_b = (const float*)d_in[5];
  const float* proj_w = (const float*)d_in[6];
  const float* proj_b = (const float*)d_in[7];
  float* out = (float*)d_out;

  char* ws = (char*)d_ws;
  auto alloc = [&](size_t bytes) -> char* {
    char* p = ws;
    ws += (bytes + 255) & ~(size_t)255;
    return p;
  };
  bhalf* xc     = (bhalf*)alloc((size_t)M2 * DD * 2);        // x || x_freq, bf16
  bhalf* wqkvc  = (bhalf*)alloc((size_t)C3 * DD * 2);
  bhalf* wprojc = (bhalf*)alloc((size_t)DD * DD * 2);
  bhalf* qkvc   = (bhalf*)alloc((size_t)M2 * C3 * 2);        // qkv both sources
  float* attn   = (float*)alloc((size_t)32 * NN * NPAD * 4); // per-b scratch
  bhalf* probs2 = (bhalf*)alloc((size_t)16 * NROWP * NPAD * 2); // per-b scratch
  bhalf* wgt    = (bhalf*)alloc((size_t)M1 * DD * 2);

  f32_to_bf16<<<512, 256, 0, stream>>>(x,  xc,                   M1 * DD);
  f32_to_bf16<<<512, 256, 0, stream>>>(xf, xc + (size_t)M1 * DD, M1 * DD);
  f32_to_bf16<<<512, 256, 0, stream>>>(qkv_w,  wqkvc,  C3 * DD);
  f32_to_bf16<<<256, 256, 0, stream>>>(proj_w, wprojc, DD * DD);

  gemm_wmma<true><<<dim3((M2 + 127) / 128, C3 / 64), 256, 0, stream>>>(
      xc, DD, M2, wqkvc, DD, qkv_b, qkvc, C3);

  const size_t smbytes = (size_t)(32 * NPAD + 16 * NPAD + 512 + 256 + 32) * sizeof(float);
  for (int b = 0; b < BB; ++b) {
    score_softmax<<<dim3(32, MTL), 128, 0, stream>>>(qkvc, attn, b);
    mix_softmax<<<dim3(NN), 256, smbytes, stream>>>(attn, conv_w, conv_b, probs2);
    weighted_gemm<<<dim3(16, MTL), 32, 0, stream>>>(probs2, qkvc, wgt, b);
  }

  gemm_wmma<false><<<dim3((M1 + 127) / 128, DD / 64), 256, 0, stream>>>(
      wgt, DD, M1, wprojc, DD, proj_b, out, DD);
}
